// AutoEncoder_14834817040831
// MI455X (gfx1250) — matrix operand order repaired and checked by reading, not compiled, not run on hardware
//
#include <hip/hip_runtime.h>
#include <hip/hip_bf16.h>
#include <math.h>

#define BB   16
#define NATC 256
#define KNN  24
#define FF   128
#define LL   4
#define EPSC 0.01f
#define NEDGE (BB*NATC*KNN)   // 98304
#define NNODE (BB*NATC)       // 4096

typedef __attribute__((ext_vector_type(16))) _Float16 v16h;
typedef __attribute__((ext_vector_type(8)))  float    v8f;

union Frag16 { v16h v; unsigned int u[8]; };

// ---- WMMA fragment loaders (wave32 16x16x32 f16 layouts per CDNA5 ISA) ----
// A: 16x32 (MxK), row-major in LDS with row stride FF halves.
__device__ inline v16h frag_a_lds(const _Float16* base, int m0, int kk, int lane) {
  int m  = m0 + (lane & 15);
  int hi = lane >> 4;
  const unsigned int* row = (const unsigned int*)(base + m * FF);
  Frag16 f;
#pragma unroll
  for (int v = 0; v < 4; ++v) f.u[v]     = row[(kk + hi*8 + 2*v) >> 1];
#pragma unroll
  for (int v = 0; v < 4; ++v) f.u[4 + v] = row[(kk + 16 + hi*8 + 2*v) >> 1];
  return f.v;
}
// B: 32x16 (KxN), stored transposed in LDS as [n][k] with row stride FF halves.
__device__ inline v16h frag_b_ldsT(const _Float16* baseT, int n0, int kk, int lane) {
  int n  = n0 + (lane & 15);
  int hi = lane >> 4;
  const unsigned int* row = (const unsigned int*)(baseT + n * FF);
  Frag16 f;
#pragma unroll
  for (int v = 0; v < 4; ++v) f.u[v]     = row[(kk + hi*8 + 2*v) >> 1];
#pragma unroll
  for (int v = 0; v < 4; ++v) f.u[4 + v] = row[(kk + 16 + hi*8 + 2*v) >> 1];
  return f.v;
}

__device__ inline float silu_f(float v) { return v * (1.0f / (1.0f + expf(-v))); }

// ---------------- setup kernels ----------------
__global__ void init_kernel(const float* __restrict__ xin, float* __restrict__ x,
                            float* __restrict__ geo, float* __restrict__ arho,
                            float* __restrict__ out, int nout) {
  int i = blockIdx.x * blockDim.x + threadIdx.x;
  if (i < nout) out[i] = 0.0f;
  if (i < NNODE*3) { float v = xin[i]; x[i] = v - floorf(v); }
  if (i < BB*9) {
    int r = i % 9;
    float v = (r == 0 || r == 4 || r == 8) ? 1.0f : 0.0f;
    geo[i] = v; arho[i] = v;
  }
}

// Transposed f16 weight copies: W1T/W2T layout [set(0=mpnn,1=upd)][l][n][k]
__global__ void convert_w_kernel(const float* __restrict__ mW1, const float* __restrict__ mW2,
                                 const float* __restrict__ uW1, const float* __restrict__ uW2,
                                 _Float16* __restrict__ W1T, _Float16* __restrict__ W2T) {
  int i = blockIdx.x * blockDim.x + threadIdx.x;
  if (i >= 2*LL*FF*FF) return;
  int k = i & 127; int n = (i >> 7) & 127; int l = (i >> 14) & 3; int s = i >> 16;
  const float* s1 = s ? uW1 : mW1;
  const float* s2 = s ? uW2 : mW2;
  W1T[i] = (_Float16)s1[(l*(FF+1) + k)*FF + n];
  W2T[i] = (_Float16)s2[(l*FF + k)*FF + n];
}

__global__ void embed_kernel(const int* __restrict__ z, const float* __restrict__ emb,
                             float* __restrict__ h, _Float16* __restrict__ hh) {
  int i = blockIdx.x * blockDim.x + threadIdx.x;
  if (i >= NNODE*FF) return;
  int f = i & 127; int bn = i >> 7;
  float v = emb[z[bn]*FF + f];
  h[i] = v; hh[i] = (_Float16)v;
}

// Periodic minimum-image kNN (cell0 == Identity in the reference)
__global__ void knn_kernel(const float* __restrict__ x, int* __restrict__ idx) {
  int bi = blockIdx.x * blockDim.x + threadIdx.x;
  if (bi >= NNODE) return;
  int b = bi / NATC, i = bi % NATC;
  float xi0 = x[bi*3], xi1 = x[bi*3+1], xi2 = x[bi*3+2];
  float bd[KNN]; int bj[KNN];
#pragma unroll
  for (int k = 0; k < KNN; ++k) { bd[k] = 1e30f; bj[k] = 0; }
  const float* xb = x + (size_t)b*NATC*3;
  for (int j = 0; j < NATC; ++j) {
    if (j == i) continue;
    float d0 = xb[j*3]   - xi0; d0 -= rintf(d0);
    float d1 = xb[j*3+1] - xi1; d1 -= rintf(d1);
    float d2 = xb[j*3+2] - xi2; d2 -= rintf(d2);
    float d = sqrtf(d0*d0 + d1*d1 + d2*d2);
    if (d < bd[KNN-1]) {
      int p = KNN - 1;
      while (p > 0 && bd[p-1] > d) { bd[p] = bd[p-1]; bj[p] = bj[p-1]; --p; }
      bd[p] = d; bj[p] = j;
    }
  }
  for (int k = 0; k < KNN; ++k) idx[bi*KNN + k] = bj[k];
}

__global__ void edge_vec_kernel(const float* __restrict__ x, const float* __restrict__ geo,
                                const int* __restrict__ idx, float* __restrict__ vec,
                                float* __restrict__ dist, float* __restrict__ u) {
  int e = blockIdx.x * blockDim.x + threadIdx.x;
  if (e >= NEDGE) return;
  int bn = e / KNN; int b = bn / NATC;
  int j = idx[e];
  const float* xi = x + (size_t)bn*3;
  const float* xj = x + ((size_t)b*NATC + j)*3;
  float f0 = xj[0]-xi[0]; f0 -= rintf(f0);
  float f1 = xj[1]-xi[1]; f1 -= rintf(f1);
  float f2 = xj[2]-xi[2]; f2 -= rintf(f2);
  const float* c = geo + b*9;
  float v0 = f0*c[0] + f1*c[3] + f2*c[6];
  float v1 = f0*c[1] + f1*c[4] + f2*c[7];
  float v2 = f0*c[2] + f1*c[5] + f2*c[8];
  float d  = sqrtf(v0*v0 + v1*v1 + v2*v2);
  vec[e*3] = v0; vec[e*3+1] = v1; vec[e*3+2] = v2;
  dist[e] = d;
  float inv = 1.0f / (d + 1e-12f);
  u[e*3] = v0*inv; u[e*3+1] = v1*inv; u[e*3+2] = v2*inv;
}

__global__ void trimask_kernel(const float* __restrict__ u, unsigned char* __restrict__ mask) {
  int e = blockIdx.x * blockDim.x + threadIdx.x;
  if (e >= NEDGE) return;
  int bn = e / KNN;
  float a0 = u[e*3], a1 = u[e*3+1], a2 = u[e*3+2];
  for (int k2 = 0; k2 < KNN; ++k2) {
    int e2 = bn*KNN + k2;
    float b0 = u[e2*3], b1 = u[e2*3+1], b2 = u[e2*3+2];
    float c0 = a1*b2 - a2*b1, c1 = a2*b0 - a0*b2, c2 = a0*b1 - a1*b0;
    mask[(size_t)e*KNN + k2] = (sqrtf(c0*c0 + c1*c1 + c2*c2) > 1e-3f) ? 1 : 0;
  }
}

// ---------------- WMMA kernels ----------------
// Fused edge GEMM + SiLU + per-atom K-reduction.
// One workgroup = 2 atoms = 48 edges; A = gathered h_j (f16), B = W1^T (f16).
__global__ __launch_bounds__(128) void edge_gemm_reduce(
    const _Float16* __restrict__ hh, const int* __restrict__ idx,
    const float* __restrict__ dist, const _Float16* __restrict__ W1T,
    const float* __restrict__ w1last, const float* __restrict__ b1,
    float* __restrict__ msum) {
  __shared__ _Float16 Alds[48*FF];
  __shared__ _Float16 Blds[FF*FF];
  __shared__ float dlds[48];
  __shared__ float msum_l[2*FF];

  int bn0 = blockIdx.x * 2;
  int b = bn0 / NATC;
  int tid = threadIdx.x;

  // gather h_j tile (48 edges x 128 f16), 4 halves per load
  for (int i = tid; i < 48*32; i += 128) {
    int e = i >> 5; int c = (i & 31) * 4;
    int a = e / KNN, k = e % KNN;
    int bn = bn0 + a;
    int j = idx[bn*KNN + k];
    *(uint2*)(Alds + e*FF + c) = *(const uint2*)(hh + ((size_t)b*NATC + j)*FF + c);
  }
  if (tid < 48) dlds[tid] = dist[(bn0 + tid/KNN)*KNN + (tid % KNN)];
  for (int i = tid; i < FF*32; i += 128) ((uint2*)Blds)[i] = ((const uint2*)W1T)[i];
  msum_l[tid] = 0.0f; msum_l[FF + tid] = 0.0f;
  __syncthreads();

  int wave = tid >> 5, lane = tid & 31;
  for (int t = wave; t < 24; t += 4) {           // 3 M-tiles x 8 N-tiles
    int m0 = (t >> 3) * 16;
    int n0 = (t & 7) * 16;
    v8f acc = {};
#pragma unroll
    for (int kk = 0; kk < FF; kk += 32) {
      v16h af = frag_a_lds(Alds, m0, kk, lane);
      v16h bf = frag_b_ldsT(Blds, n0, kk, lane);
      acc = __builtin_amdgcn_wmma_f32_16x16x32_f16(false, af, false, bf,
                                                   (short)0, acc, false, false);
    }
    int hi = lane >> 4;
    int nn = n0 + (lane & 15);
    float wl = w1last[nn], bb = b1[nn];
#pragma unroll
    for (int r = 0; r < 8; ++r) {
      int row = m0 + r + hi*8;                   // edge row 0..47
      float v = acc[r] + dlds[row]*wl + bb;
      v = silu_f(v);
      atomicAdd(&msum_l[(row / KNN)*FF + nn], v);  // per-atom segmented sum
    }
  }
  __syncthreads();
  msum[(size_t)bn0*FF + tid]      = msum_l[tid];
  msum[(size_t)bn0*FF + FF + tid] = msum_l[FF + tid];
}

// h += silu(msum @ W2 + b2); refresh f16 mirror. One workgroup = 16 atoms.
__global__ __launch_bounds__(128) void node_gemm(
    const float* __restrict__ msum, const _Float16* __restrict__ W2T,
    const float* __restrict__ b2, float* __restrict__ h, _Float16* __restrict__ hh) {
  __shared__ _Float16 Alds[16*FF];
  __shared__ _Float16 Blds[FF*FF];
  int row0 = blockIdx.x * 16;
  int tid = threadIdx.x;
  for (int i = tid; i < 16*FF; i += 128) Alds[i] = (_Float16)msum[(size_t)row0*FF + i];
  for (int i = tid; i < FF*32; i += 128) ((uint2*)Blds)[i] = ((const uint2*)W2T)[i];
  __syncthreads();
  int wave = tid >> 5, lane = tid & 31;
#pragma unroll
  for (int nt = 0; nt < 2; ++nt) {
    int n0 = wave*32 + nt*16;
    v8f acc = {};
#pragma unroll
    for (int kk = 0; kk < FF; kk += 32) {
      v16h af = frag_a_lds(Alds, 0, kk, lane);
      v16h bf = frag_b_ldsT(Blds, n0, kk, lane);
      acc = __builtin_amdgcn_wmma_f32_16x16x32_f16(false, af, false, bf,
                                                   (short)0, acc, false, false);
    }
    int hi = lane >> 4;
    int nn = n0 + (lane & 15);
    float bb = b2[nn];
#pragma unroll
    for (int r = 0; r < 8; ++r) {
      int row = row0 + r + hi*8;
      float v = silu_f(acc[r] + bb);
      float nh = h[(size_t)row*FF + nn] + v;
      h[(size_t)row*FF + nn] = nh;
      hh[(size_t)row*FF + nn] = (_Float16)nh;
    }
  }
}

// ---------------- update-phase scalar kernels ----------------
__global__ void edgedot_kernel(const float* __restrict__ h, const int* __restrict__ idx,
                               const float* __restrict__ wev, const float* __restrict__ wpv,
                               float* __restrict__ w_e, float* __restrict__ w_p) {
  int e = blockIdx.x * blockDim.x + threadIdx.x;
  if (e >= NEDGE) return;
  int bn = e / KNN; int b = bn / NATC;
  int j = idx[e];
  const float* hi = h + (size_t)bn*FF;
  const float* hj = h + ((size_t)b*NATC + j)*FF;
  float s1 = 0.0f, s2 = 0.0f;
  for (int f = 0; f < FF; ++f) {
    float ef = hi[f] + hj[f];
    s1 += ef * wev[f];
    s2 += ef * wpv[f];
  }
  w_e[e] = tanhf(s1);
  w_p[e] = tanhf(s2);
}

__global__ void zero_accum_kernel(float* __restrict__ accum) {
  int i = threadIdx.x;
  if (i < BB*18) accum[i] = 0.0f;
}

__global__ __launch_bounds__(64) void strain_tri_kernel(
    const float* __restrict__ u, const float* __restrict__ w_e,
    const unsigned char* __restrict__ mask, float* __restrict__ accum) {
  int bn = blockIdx.x; int b = bn / NATC;
  int t = threadIdx.x;
  __shared__ float su[KNN][3];
  __shared__ float sw[KNN];
  __shared__ float sacc[18];
  if (t < KNN) {
    su[t][0] = u[(bn*KNN+t)*3]; su[t][1] = u[(bn*KNN+t)*3+1]; su[t][2] = u[(bn*KNN+t)*3+2];
    sw[t] = w_e[bn*KNN + t];
  }
  if (t < 18) sacc[t] = 0.0f;
  __syncthreads();
  float loc[18];
#pragma unroll
  for (int i = 0; i < 18; ++i) loc[i] = 0.0f;
  if (t < KNN) {
    float w = sw[t];
    float a0 = su[t][0], a1 = su[t][1], a2 = su[t][2];
    loc[0] = w*a0*a0; loc[1] = w*a0*a1; loc[2] = w*a0*a2;
    loc[3] = w*a1*a0; loc[4] = w*a1*a1; loc[5] = w*a1*a2;
    loc[6] = w*a2*a0; loc[7] = w*a2*a1; loc[8] = w*a2*a2;
  }
  for (int p = t; p < KNN*KNN; p += 64) {
    int k = p / KNN, k2 = p % KNN;
    if (mask[((size_t)bn*KNN + k)*KNN + k2]) {
      float w = sw[k]*sw[k2];
      float a0 = su[k][0], a1 = su[k][1], a2 = su[k][2];
      float b0 = su[k2][0], b1 = su[k2][1], b2 = su[k2][2];
      float c0 = a1*b2 - a2*b1, c1 = a2*b0 - a0*b2, c2 = a0*b1 - a1*b0;
      loc[9]  += w*c0*c0; loc[10] += w*c0*c1; loc[11] += w*c0*c2;
      loc[12] += w*c1*c0; loc[13] += w*c1*c1; loc[14] += w*c1*c2;
      loc[15] += w*c2*c0; loc[16] += w*c2*c1; loc[17] += w*c2*c2;
    }
  }
#pragma unroll
  for (int i = 0; i < 18; ++i) if (loc[i] != 0.0f) atomicAdd(&sacc[i], loc[i]);
  __syncthreads();
  if (t < 18) atomicAdd(&accum[b*18 + t], sacc[t]);
}

__global__ void matrix_update_kernel(const float* __restrict__ accum, float* __restrict__ geo,
                                     float* __restrict__ arho, float* __restrict__ invc,
                                     float* __restrict__ out_rho) {
  int b = threadIdx.x;
  if (b >= BB) return;
  const float* ac = accum + b*18;
  float A[9];
#pragma unroll
  for (int i = 0; i < 9; ++i) {
    float s  = ac[i]   * (1.0f / (float)(NATC*KNN));
    float tr = ac[9+i] * (1.0f / (float)(NATC*KNN*KNN));
    A[i] = EPSC*(s + tr) + ((i == 0 || i == 4 || i == 8) ? 1.0f : 0.0f);
  }
  float g[9];
#pragma unroll
  for (int i = 0; i < 9; ++i) g[i] = geo[b*9 + i];
  float det = g[0]*(g[4]*g[8]-g[5]*g[7]) - g[1]*(g[3]*g[8]-g[5]*g[6]) + g[2]*(g[3]*g[7]-g[4]*g[6]);
  float id = 1.0f / det;
  float iv[9];
  iv[0] = (g[4]*g[8]-g[5]*g[7])*id; iv[1] = (g[2]*g[7]-g[1]*g[8])*id; iv[2] = (g[1]*g[5]-g[2]*g[4])*id;
  iv[3] = (g[5]*g[6]-g[3]*g[8])*id; iv[4] = (g[0]*g[8]-g[2]*g[6])*id; iv[5] = (g[2]*g[3]-g[0]*g[5])*id;
  iv[6] = (g[3]*g[7]-g[4]*g[6])*id; iv[7] = (g[1]*g[6]-g[0]*g[7])*id; iv[8] = (g[0]*g[4]-g[1]*g[3])*id;
#pragma unroll
  for (int i = 0; i < 9; ++i) invc[b*9 + i] = iv[i];
  float old[9], r[9];
#pragma unroll
  for (int i = 0; i < 9; ++i) old[i] = arho[b*9 + i];
#pragma unroll
  for (int i = 0; i < 3; ++i)
#pragma unroll
    for (int j = 0; j < 3; ++j)
      r[i*3+j] = A[i*3+0]*old[0*3+j] + A[i*3+1]*old[1*3+j] + A[i*3+2]*old[2*3+j];
#pragma unroll
  for (int i = 0; i < 9; ++i) {
    arho[b*9 + i] = r[i];
    geo[b*9 + i]  = r[i];     // rho_prime = action_rho @ I
    out_rho[b*9 + i] = r[i];  // last layer's value is the final output
  }
}

__global__ void pos_update_kernel(const float* __restrict__ w_p, const float* __restrict__ vec,
                                  const float* __restrict__ invc, float* __restrict__ x,
                                  float* __restrict__ traj) {
  int bn = blockIdx.x * blockDim.x + threadIdx.x;
  if (bn >= NNODE) return;
  int b = bn / NATC;
  float c0 = 0.0f, c1 = 0.0f, c2 = 0.0f;
#pragma unroll
  for (int k = 0; k < KNN; ++k) {
    int e = bn*KNN + k;
    float w = w_p[e];
    c0 += w*vec[e*3]; c1 += w*vec[e*3+1]; c2 += w*vec[e*3+2];
  }
  c0 *= EPSC; c1 *= EPSC; c2 *= EPSC;
  traj[bn*3]   += c0; traj[bn*3+1] += c1; traj[bn*3+2] += c2;
  const float* iv = invc + b*9;
  x[bn*3]   += c0*iv[0] + c1*iv[3] + c2*iv[6];
  x[bn*3+1] += c0*iv[1] + c1*iv[4] + c2*iv[7];
  x[bn*3+2] += c0*iv[2] + c1*iv[5] + c2*iv[8];
}

// ---------------- launch ----------------
extern "C" void kernel_launch(void* const* d_in, const int* in_sizes, int n_in,
                              void* d_out, int out_size, void* d_ws, size_t ws_size,
                              hipStream_t stream) {
  (void)in_sizes; (void)n_in; (void)ws_size;
  const float* in_x   = (const float*)d_in[1];
  const int*   in_z   = (const int*)d_in[2];
  const float* emb    = (const float*)d_in[4];
  const float* mW1    = (const float*)d_in[5];
  const float* mb1    = (const float*)d_in[6];
  const float* mW2    = (const float*)d_in[7];
  const float* mb2    = (const float*)d_in[8];
  const float* uW1    = (const float*)d_in[9];
  const float* ub1    = (const float*)d_in[10];
  const float* uW2    = (const float*)d_in[11];
  const float* ub2    = (const float*)d_in[12];
  const float* actwe  = (const float*)d_in[13];
  const float* actpwe = (const float*)d_in[14];

  char* p = (char*)d_ws;
  auto alloc = [&](size_t bytes) -> void* {
    void* r = (void*)p; p += (bytes + 255) & ~(size_t)255; return r;
  };
  float*         x     = (float*)alloc((size_t)NNODE*3*4);
  int*           idx   = (int*)alloc((size_t)NEDGE*4);
  float*         vec   = (float*)alloc((size_t)NEDGE*3*4);
  float*         dist  = (float*)alloc((size_t)NEDGE*4);
  float*         u     = (float*)alloc((size_t)NEDGE*3*4);
  unsigned char* tmask = (unsigned char*)alloc((size_t)NEDGE*KNN);
  float*         h     = (float*)alloc((size_t)NNODE*FF*4);
  _Float16*      hh    = (_Float16*)alloc((size_t)NNODE*FF*2);
  float*         msum  = (float*)alloc((size_t)NNODE*FF*4);
  _Float16*      W1T   = (_Float16*)alloc((size_t)2*LL*FF*FF*2);
  _Float16*      W2T   = (_Float16*)alloc((size_t)2*LL*FF*FF*2);
  float*         w_e   = (float*)alloc((size_t)NEDGE*4);
  float*         w_p   = (float*)alloc((size_t)NEDGE*4);
  float*         accum = (float*)alloc((size_t)BB*18*4);
  float*         geo   = (float*)alloc((size_t)BB*9*4);
  float*         arho  = (float*)alloc((size_t)BB*9*4);
  float*         invc  = (float*)alloc((size_t)BB*9*4);

  float* traj    = (float*)d_out;
  float* out_rho = traj + (size_t)NNODE*3;

  init_kernel<<<64, 256, 0, stream>>>(in_x, x, geo, arho, traj, out_size);
  convert_w_kernel<<<512, 256, 0, stream>>>(mW1, mW2, uW1, uW2, W1T, W2T);
  embed_kernel<<<2048, 256, 0, stream>>>(in_z, emb, h, hh);
  knn_kernel<<<16, 256, 0, stream>>>(x, idx);
  edge_vec_kernel<<<384, 256, 0, stream>>>(x, geo, idx, vec, dist, u);
  trimask_kernel<<<384, 256, 0, stream>>>(u, tmask);

  // message-passing layers
  for (int l = 0; l < LL; ++l) {
    edge_gemm_reduce<<<NNODE/2, 128, 0, stream>>>(
        hh, idx, dist, W1T + (size_t)l*FF*FF,
        mW1 + ((size_t)l*(FF+1) + FF)*FF, mb1 + (size_t)l*FF, msum);
    node_gemm<<<NNODE/16, 128, 0, stream>>>(
        msum, W2T + (size_t)l*FF*FF, mb2 + (size_t)l*FF, h, hh);
  }
  // update layers
  for (int l = 0; l < LL; ++l) {
    edge_gemm_reduce<<<NNODE/2, 128, 0, stream>>>(
        hh, idx, dist, W1T + (size_t)(LL + l)*FF*FF,
        uW1 + ((size_t)l*(FF+1) + FF)*FF, ub1 + (size_t)l*FF, msum);
    node_gemm<<<NNODE/16, 128, 0, stream>>>(
        msum, W2T + (size_t)(LL + l)*FF*FF, ub2 + (size_t)l*FF, h, hh);
    edgedot_kernel<<<384, 256, 0, stream>>>(h, idx, actwe + (size_t)l*FF,
                                            actpwe + (size_t)l*FF, w_e, w_p);
    zero_accum_kernel<<<1, 512, 0, stream>>>(accum);
    strain_tri_kernel<<<NNODE, 64, 0, stream>>>(u, w_e, tmask, accum);
    matrix_update_kernel<<<1, 32, 0, stream>>>(accum, geo, arho, invc, out_rho);
    pos_update_kernel<<<16, 256, 0, stream>>>(w_p, vec, invc, x, traj);
    edge_vec_kernel<<<384, 256, 0, stream>>>(x, geo, idx, vec, dist, u);
  }
}
